// WeightedSAGE_14474039787720
// MI455X (gfx1250) — matrix operand order repaired and checked by reading, not compiled, not run on hardware
//
#include <hip/hip_runtime.h>
#include <hip/hip_bf16.h>

typedef __bf16 bf16_t;
typedef __bf16 v16bf __attribute__((ext_vector_type(16)));
typedef float  v8f   __attribute__((ext_vector_type(8)));

#define NN 50000
#define NE 800000
#define D  128
#define KT 256          // concat(h, agg) depth
#define LDSW 132        // padded LDS row stride in u32 (264 bf16) to dodge bank conflicts

// ---------------- f32 -> bf16 conversion ----------------
__global__ void wsage_cvt_bf16(const float* __restrict__ in,
                               bf16_t* __restrict__ out, int n) {
    int i = (blockIdx.x * blockDim.x + threadIdx.x) * 4;
    if (i + 3 < n) {
        float4 v = *(const float4*)(in + i);
        out[i + 0] = (bf16_t)v.x;
        out[i + 1] = (bf16_t)v.y;
        out[i + 2] = (bf16_t)v.z;
        out[i + 3] = (bf16_t)v.w;
    }
}

// ---------------- zero f32 buffer (float4 stores) ----------------
__global__ void wsage_zero_f32(float* __restrict__ p, int n4) {
    int i = blockIdx.x * blockDim.x + threadIdx.x;
    if (i < n4) {
        float4 z; z.x = 0.f; z.y = 0.f; z.z = 0.f; z.w = 0.f;
        ((float4*)p)[i] = z;
    }
}

// ---------------- edge scatter: agg[dst] += w * h[src] ----------------
// one wave (32 lanes) per edge, 4 feature dims per lane, f32 HW atomics
__global__ __launch_bounds__(256)
void wsage_scatter(const bf16_t* __restrict__ h,
                   const int* __restrict__ src,
                   const int* __restrict__ dst,
                   const float* __restrict__ w,
                   float* __restrict__ agg) {
    int t    = blockIdx.x * blockDim.x + threadIdx.x;
    int e    = t >> 5;
    int lane = t & 31;
    if (e >= NE) return;
    int   s  = src[e];
    int   dd = dst[e];
    float we = w[e];
    int base = lane * 4;
    union { uint2 u; bf16_t b[4]; } hv;
    hv.u = *(const uint2*)(h + (size_t)s * D + base);
    float* ap = agg + (size_t)dd * D + base;
    unsafeAtomicAdd(ap + 0, (float)hv.b[0] * we);
    unsafeAtomicAdd(ap + 1, (float)hv.b[1] * we);
    unsafeAtomicAdd(ap + 2, (float)hv.b[2] * we);
    unsafeAtomicAdd(ap + 3, (float)hv.b[3] * we);
}

// ---------------- fused concat-GEMM + bias (+ReLU) via WMMA bf16 ----------------
// block = 256 threads = 8 waves; block owns 16 nodes; wave owns 16 output cols.
// out = act( [h | agg] @ W + b ),  W is bf16 [256][128] row-major.
__global__ __launch_bounds__(256)
void wsage_gemm(const bf16_t* __restrict__ h,     // [NN][128] bf16
                const float*  __restrict__ agg,   // [NN][128] f32
                const bf16_t* __restrict__ Wb,    // [256][128] bf16
                const float*  __restrict__ bias,  // [128] f32
                bf16_t* __restrict__ out_bf,      // next-layer h (if !last)
                float*  __restrict__ out_f32,     // final output (if last)
                int is_last) {
    __shared__ bf16_t Atile[16 * 2 * LDSW];       // 16 rows x 264 bf16 (256 used)
    uint* A32 = (uint*)Atile;

    int tid   = threadIdx.x;
    int node0 = blockIdx.x * 16;

    // stage h part: 16 rows x 64 u32 (128 bf16) straight copy
    const uint* H32 = (const uint*)(h + (size_t)node0 * D);
    for (int i = tid; i < 16 * 64; i += 256) {
        int r = i >> 6, c = i & 63;
        A32[r * LDSW + c] = H32[i];
    }
    // stage agg part: f32 -> bf16 pairs packed into u32
    const float* G = agg + (size_t)node0 * D;
    for (int i = tid; i < 16 * 64; i += 256) {
        int r = i >> 6, c = i & 63;
        float f0 = G[r * D + c * 2 + 0];
        float f1 = G[r * D + c * 2 + 1];
        union { bf16_t b[2]; uint u; } pk;
        pk.b[0] = (bf16_t)f0;
        pk.b[1] = (bf16_t)f1;
        A32[r * LDSW + 64 + c] = pk.u;
    }
    __syncthreads();

    int wave  = tid >> 5;         // 0..7 -> output column tile
    int lane  = tid & 31;
    int half  = lane >> 4;        // lane-group within wave
    int m     = lane & 15;        // A-matrix row this lane contributes
    int ncol0 = wave * 16;

    v8f acc = {};
    #pragma unroll
    for (int ks = 0; ks < 8; ++ks) {
        // ---- A fragment (16x32 bf16): lane=M; VGPR v holds K pair at
        // base = (v>>2)*16 + half*8 + (v&3)*2  (ISA 16-bit A layout)
        union { uint u[8]; v16bf v; } af;
        #pragma unroll
        for (int v = 0; v < 8; ++v) {
            int kb = ks * 32 + (v >> 2) * 16 + half * 8 + (v & 3) * 2;
            af.u[v] = A32[m * LDSW + (kb >> 1)];
        }
        // ---- B fragment (32x16 bf16): lane = N column (ncol0 + lane&15), element s holds
        // K = (s>>3)*16 + half*8 + (s&7)  (same K order as A)
        union { bf16_t e[16]; v16bf v; } bfr;
        #pragma unroll
        for (int s = 0; s < 16; ++s) {
            int krow = ks * 32 + (s >> 3) * 16 + half * 8 + (s & 7);
            bfr.e[s] = Wb[(size_t)krow * D + ncol0 + m];
        }

        acc = __builtin_amdgcn_wmma_f32_16x16x32_bf16(
                  false, af.v, false, bfr.v, (short)0, acc, false, false);
    }

    // ---- C/D layout: lane -> N = lane&15, VGPR j -> M = half*8 + j
    int col = ncol0 + (lane & 15);
    float bv = bias[col];
    #pragma unroll
    for (int j = 0; j < 8; ++j) {
        int row = half * 8 + j;
        float val = acc[j] + bv;
        if (is_last) {
            out_f32[(size_t)(node0 + row) * D + col] = val;
        } else {
            val = val > 0.f ? val : 0.f;
            out_bf[(size_t)(node0 + row) * D + col] = (bf16_t)val;
        }
    }
}

extern "C" void kernel_launch(void* const* d_in, const int* in_sizes, int n_in,
                              void* d_out, int out_size, void* d_ws, size_t ws_size,
                              hipStream_t stream) {
    const float* x   = (const float*)d_in[0];
    const int*   src = (const int*)  d_in[1];
    const int*   dst = (const int*)  d_in[2];
    const float* w   = (const float*)d_in[3];
    const float* Wf[4] = { (const float*)d_in[4], (const float*)d_in[6],
                           (const float*)d_in[8], (const float*)d_in[10] };
    const float* bv[4] = { (const float*)d_in[5], (const float*)d_in[7],
                           (const float*)d_in[9], (const float*)d_in[11] };

    // workspace layout
    char*  ws = (char*)d_ws;
    size_t o  = 0;
    bf16_t* hA  = (bf16_t*)(ws + o); o += (size_t)NN * D * sizeof(bf16_t); o = (o + 255) & ~(size_t)255;
    bf16_t* hB  = (bf16_t*)(ws + o); o += (size_t)NN * D * sizeof(bf16_t); o = (o + 255) & ~(size_t)255;
    float*  agg = (float*) (ws + o); o += (size_t)NN * D * sizeof(float);  o = (o + 255) & ~(size_t)255;
    bf16_t* Wb  = (bf16_t*)(ws + o); o += (size_t)4 * KT * D * sizeof(bf16_t);

    // convert x and all four weight matrices to bf16
    {
        int n = NN * D;
        wsage_cvt_bf16<<<(n / 4 + 255) / 256, 256, 0, stream>>>(x, hA, n);
        int nw = KT * D;
        for (int l = 0; l < 4; ++l)
            wsage_cvt_bf16<<<(nw / 4 + 255) / 256, 256, 0, stream>>>(Wf[l], Wb + (size_t)l * KT * D, nw);
    }

    bf16_t* cur = hA;
    bf16_t* nxt = hB;
    int n4 = NN * D / 4;
    for (int l = 0; l < 4; ++l) {
        wsage_zero_f32<<<(n4 + 255) / 256, 256, 0, stream>>>(agg, n4);
        wsage_scatter<<<(NE * 32) / 256, 256, 0, stream>>>(cur, src, dst, w, agg);
        int last = (l == 3);
        wsage_gemm<<<NN / 16, 256, 0, stream>>>(cur, agg,
                                                Wb + (size_t)l * KT * D, bv[l],
                                                nxt, (float*)d_out, last);
        bf16_t* t = cur; cur = nxt; nxt = t;
    }
}
